// Block_73443940761664
// MI455X (gfx1250) — matrix operand order repaired and checked by reading, not compiled, not run on hardware
//
#include <hip/hip_runtime.h>
#include <hip/hip_bf16.h>
#include <math.h>

typedef __attribute__((ext_vector_type(16))) __bf16 v16bf;
typedef __attribute__((ext_vector_type(8)))  float  v8f;
typedef __attribute__((ext_vector_type(4))) unsigned int u32x4;
typedef __attribute__((ext_vector_type(8))) int          i32x8;
typedef __attribute__((ext_vector_type(4))) int          i32x4;

#define TILE   128
#define TKS    32
#define APITCH 40   // bf16 elements per LDS row (32 data + 8 pad = 80 B pitch)
#define PQ     72   // attention LDS pitch (64 + 8 pad)

#if defined(__has_builtin)
#if __has_builtin(__builtin_amdgcn_tensor_load_to_lds)
#define HAVE_TDM 1
#endif
#endif

static __device__ __forceinline__ unsigned short f2bf(float f) {
  unsigned int u = __float_as_uint(f);
  u += 0x7FFFu + ((u >> 16) & 1u);
  return (unsigned short)(u >> 16);
}
static __device__ __forceinline__ float bf2f(unsigned short h) {
  return __uint_as_float(((unsigned int)h) << 16);
}

struct U4x2 { uint4 a, b; };
static __device__ __forceinline__ v16bf frag_ld2(const unsigned short* p0,
                                                 const unsigned short* p1) {
  U4x2 t;
  t.a = *(const uint4*)p0;
  t.b = *(const uint4*)p1;
  return __builtin_bit_cast(v16bf, t);
}
static __device__ __forceinline__ v8f vzero8() {
  v8f v;
#pragma unroll
  for (int i = 0; i < 8; ++i) v[i] = 0.0f;
  return v;
}
#define WMMA_BF16(A_, B_, C_) \
  __builtin_amdgcn_wmma_f32_16x16x32_bf16(false, (A_), false, (B_), (short)0, (C_), false, false)

static __device__ __forceinline__ unsigned lds_off(const void* p) {
  // generic-space LDS pointer: low 32 bits are the wave-relative LDS byte addr
  return (unsigned)(unsigned long long)p;
}

#ifdef HAVE_TDM
// Issue a TDM 2D tile load: tileY rows x tileX bf16 elements, global row
// stride strideElems, into LDS with 16 B pad after each 64 B row (pitch 80 B).
static __device__ __forceinline__ void tdm_load_tile_bf16(
    unsigned lds_addr, unsigned long long gaddr, unsigned strideElems,
    unsigned tileX, unsigned tileY) {
  u32x4 g0;
  g0[0] = 1u;  // count=1, is_restore=0, gather off
  g0[1] = lds_addr;
  g0[2] = (unsigned)(gaddr & 0xFFFFFFFFull);
  g0[3] = (unsigned)((gaddr >> 32) & 0x1FFFFFFull) | 0x80000000u;  // type=2
  i32x8 g1;
  const unsigned w0 = (1u << 16)   // data_size = 2 bytes
                    | (1u << 20)   // pad_enable
                    | (3u << 22)   // pad_interval: 16 DWORDs (= one 64 B row)
                    | (3u << 25);  // pad_amount: 4 DWORDs (= 16 B)
  g1[0] = (int)w0;
  g1[1] = (int)((tileX & 0xFFFFu) << 16);                     // tensor_dim0 lo16
  g1[2] = (int)((tileX >> 16) | ((tileY & 0xFFFFu) << 16));   // dim0 hi | dim1 lo
  g1[3] = (int)((tileY >> 16) | (tileX << 16));               // dim1 hi | tile_dim0
  g1[4] = (int)(tileY & 0xFFFFu);                             // tile_dim1, tile_dim2=0
  g1[5] = (int)strideElems;                                   // tensor_dim0_stride lo32
  g1[6] = 0;
  g1[7] = 0;
  i32x4 z4;
  z4[0] = 0; z4[1] = 0; z4[2] = 0; z4[3] = 0;
  i32x8 z8;
#pragma unroll
  for (int i = 0; i < 8; ++i) z8[i] = 0;
  __builtin_amdgcn_tensor_load_to_lds(g0, g1, z4, z4, z8, 0);
}
#endif

// ---------------------------------------------------------------------------
// Transpose + fp32->bf16 weight conversion: src [z][K][N] f32 -> dst [z][N][K]
// ---------------------------------------------------------------------------
__global__ void __launch_bounds__(256) transpose_cvt_kernel(
    const float* __restrict__ src, unsigned short* __restrict__ dst,
    int K, int N) {
  __shared__ float t[32][33];
  const long zoff = (long)blockIdx.z * K * N;
  const int tx = threadIdx.x & 31;
  const int ty = threadIdx.x >> 5;  // 0..7
  const int n0 = blockIdx.x * 32, k0 = blockIdx.y * 32;
#pragma unroll
  for (int j = 0; j < 4; ++j)
    t[ty + j * 8][tx] = src[zoff + (long)(k0 + ty + j * 8) * N + n0 + tx];
  __syncthreads();
#pragma unroll
  for (int j = 0; j < 4; ++j)
    dst[zoff + (long)(n0 + ty + j * 8) * K + k0 + tx] = f2bf(t[tx][ty + j * 8]);
}

// ---------------------------------------------------------------------------
// LayerNorm: one block per token, fp32 in, bf16 out
// ---------------------------------------------------------------------------
__global__ void __launch_bounds__(256) ln_kernel(const float* __restrict__ x,
                                                 const float* __restrict__ g,
                                                 const float* __restrict__ b,
                                                 unsigned short* __restrict__ out,
                                                 int D) {
  __shared__ float red[256];
  const int tid = threadIdx.x;
  const long t = blockIdx.x;
  const float* xr = x + t * D;

  float s = 0.f;
  for (int i = tid; i < D; i += 256) s += xr[i];
  red[tid] = s;
  __syncthreads();
  for (int st = 128; st > 0; st >>= 1) {
    if (tid < st) red[tid] += red[tid + st];
    __syncthreads();
  }
  const float mean = red[0] / (float)D;
  __syncthreads();

  float vs = 0.f;
  for (int i = tid; i < D; i += 256) {
    float d = xr[i] - mean;
    vs += d * d;
  }
  red[tid] = vs;
  __syncthreads();
  for (int st = 128; st > 0; st >>= 1) {
    if (tid < st) red[tid] += red[tid + st];
    __syncthreads();
  }
  const float inv = rsqrtf(red[0] / (float)D + 1e-5f);

  for (int i = tid; i < D; i += 256)
    out[t * D + i] = f2bf((xr[i] - mean) * inv * g[i] + b[i]);
}

// ---------------------------------------------------------------------------
// bf16 WMMA GEMM: C[M,N] = A[M,K](bf16) * Bt[N,K](bf16, pre-transposed) + bias
// B tiles (and A tiles when not gathering) are DMA'd by the Tensor Data Mover
// into double-buffered LDS; wave 0 issues, s_wait_tensorcnt + barrier syncs.
//   epi 0: bf16 out = acc + bias
//   epi 1: f32 out  = acc + bias + residual
//   epi 2: bf16 out = gelu(acc + bias)
//   epi 3: f32 out  = (acc + bias) * rowScale[row]
// ---------------------------------------------------------------------------
__global__ void __launch_bounds__(256) gemm_bf16_kernel(
    const unsigned short* __restrict__ A, long lda,
    const unsigned short* __restrict__ Bt, long ldbK, long bStride,
    const float* __restrict__ biasPtr, long biasStride,
    const float* __restrict__ residual,
    const int* __restrict__ segBase, const int* __restrict__ segCount,
    const int* __restrict__ gatherIdx,
    const float* __restrict__ rowScale,
    float* __restrict__ outF, unsigned short* __restrict__ outBF,
    long outStride, int M, int N, int K, int epi) {
  __shared__ unsigned short lA[2][TILE * APITCH];
  __shared__ unsigned short lB[2][TILE * APITCH];

  const int tid = threadIdx.x;
  const int z = blockIdx.z;
  const int tM = blockIdx.y, tN = blockIdx.x;
  const int seg0 = segBase ? segBase[z] : 0;
  const int rowsSeg = segCount ? segCount[z] : M;
  const int valid = rowsSeg - tM * TILE;
  if (valid <= 0) return;
  const int vclamp = (valid < TILE ? valid : TILE) - 1;

  const unsigned short* BtZ = Bt + (long)z * bStride + (long)tN * TILE * ldbK;
  const float* bias = biasPtr + (long)z * biasStride;

  // manual A staging (gather path)
  const int sRow = tid >> 1;
  const int sHalf = tid & 1;
  const int rloc = sRow <= vclamp ? sRow : vclamp;
  const int segIdxA = seg0 + tM * TILE + rloc;
  const int srcRow = gatherIdx ? gatherIdx[segIdxA] : segIdxA;
  const unsigned short* aSrc = A + (long)srcRow * lda + sHalf * 16;
  // TDM A base (contiguous-rows path)
  const unsigned long long aTileBase =
      (unsigned long long)(A + (long)(seg0 + (long)tM * TILE) * lda);

  const int lane = tid & 31;
  const int wave = tid >> 5;
  const int wM = (wave >> 1) * 32;
  const int wN = (wave & 1) * 64;
  const int lr = lane & 15;
  const int hi = lane >> 4;

#ifdef HAVE_TDM
  const bool tdmA = (gatherIdx == nullptr);
#else
  const bool tdmA = false;
#endif

  auto stageB = [&](int k0, int buf) {
#ifdef HAVE_TDM
    if (tid < 32)
      tdm_load_tile_bf16(lds_off(&lB[buf][0]),
                         (unsigned long long)BtZ + (unsigned long long)k0 * 2ull,
                         (unsigned)ldbK, TKS, TILE);
#else
    const uint4* p = (const uint4*)(BtZ + (long)sRow * ldbK + k0 + sHalf * 16);
    uint4 x0 = p[0], x1 = p[1];
    uint4* lp = (uint4*)(&lB[buf][sRow * APITCH + sHalf * 16]);
    lp[0] = x0; lp[1] = x1;
#endif
  };
  auto stageA = [&](int k0, int buf) {
#ifdef HAVE_TDM
    if (tdmA) {
      if (tid < 32)
        tdm_load_tile_bf16(lds_off(&lA[buf][0]),
                           aTileBase + (unsigned long long)k0 * 2ull,
                           (unsigned)lda, TKS, TILE);
      return;
    }
#endif
    const uint4* p = (const uint4*)(aSrc + k0);
    uint4 x0 = p[0], x1 = p[1];
    uint4* lp = (uint4*)(&lA[buf][sRow * APITCH + sHalf * 16]);
    lp[0] = x0; lp[1] = x1;
  };

  v8f acc[2][4];
#pragma unroll
  for (int i = 0; i < 2; ++i)
#pragma unroll
    for (int j = 0; j < 4; ++j) acc[i][j] = vzero8();

  stageB(0, 0);
  stageA(0, 0);
#ifdef HAVE_TDM
  if (tid < 32) __builtin_amdgcn_s_wait_tensorcnt(0);
#endif
  __syncthreads();

  int p = 0;
  for (int k0 = 0; k0 < K; k0 += TKS) {
    const int nxt = p ^ 1;
    const bool more = (k0 + TKS) < K;
    if (more) {
      stageB(k0 + TKS, nxt);
      stageA(k0 + TKS, nxt);
      if (!tdmA) __builtin_prefetch(aSrc + k0 + 2 * TKS, 0, 1);
    }

    v16bf af[2], bfr[4];
#pragma unroll
    for (int i = 0; i < 2; ++i) {
      const unsigned short* pp = &lA[p][(wM + i * 16 + lr) * APITCH + hi * 8];
      af[i] = frag_ld2(pp, pp + 16);
    }
#pragma unroll
    for (int j = 0; j < 4; ++j) {
      const unsigned short* pp = &lB[p][(wN + j * 16 + lr) * APITCH + hi * 8];
      bfr[j] = frag_ld2(pp, pp + 16);
    }
#pragma unroll
    for (int i = 0; i < 2; ++i)
#pragma unroll
      for (int j = 0; j < 4; ++j) acc[i][j] = WMMA_BF16(af[i], bfr[j], acc[i][j]);

#ifdef HAVE_TDM
    if (more && tid < 32) __builtin_amdgcn_s_wait_tensorcnt(0);
#endif
    __syncthreads();
    p = nxt;
  }

  // epilogue: C layout — VGPR r holds row (r + 8*hi), col = lane&15
#pragma unroll
  for (int i = 0; i < 2; ++i) {
    const int mloc0 = wM + i * 16 + hi * 8;
#pragma unroll
    for (int j = 0; j < 4; ++j) {
      const int colG = tN * TILE + wN + j * 16 + lr;
      const float bv = bias[colG];
#pragma unroll
      for (int r = 0; r < 8; ++r) {
        const int lRow = mloc0 + r;
        if (lRow > vclamp) continue;
        const long outRow = seg0 + (long)tM * TILE + lRow;
        float v = acc[i][j][r] + bv;
        if (epi == 0) {
          outBF[outRow * outStride + colG] = f2bf(v);
        } else if (epi == 1) {
          outF[outRow * outStride + colG] = v + residual[outRow * outStride + colG];
        } else if (epi == 2) {
          float gl = 0.5f * v * (1.0f + erff(v * 0.70710678f));
          outBF[outRow * outStride + colG] = f2bf(gl);
        } else {
          outF[outRow * outStride + colG] = v * rowScale[outRow];
        }
      }
    }
  }
}

// ---------------------------------------------------------------------------
// Flash attention: one workgroup = (head, 128 queries), 8 waves x 16 queries.
// QK^T and P*V both via WMMA; online softmax in registers + 16-lane shuffles.
// ---------------------------------------------------------------------------
__global__ void __launch_bounds__(256) attn_kernel(
    const unsigned short* __restrict__ Qb, const unsigned short* __restrict__ Kb,
    const unsigned short* __restrict__ Vb, unsigned short* __restrict__ Ob,
    int S, int D) {
  __shared__ unsigned short sQ[128 * PQ];
  __shared__ unsigned short sK[64 * PQ];
  __shared__ unsigned short sVt[64 * PQ];
  __shared__ unsigned short sP[128 * PQ];

  const int tid = threadIdx.x;
  const int h = blockIdx.y;
  const int qBase = blockIdx.x * 128;
  const int lane = tid & 31, wave = tid >> 5;
  const int lr = lane & 15, hi = lane >> 4;
  const int hcol = h * 64;
  const int mw = wave * 16;

  {  // stage Q tile [128][64]
    const int row = tid >> 1, half = tid & 1;
    const uint4* p = (const uint4*)(Qb + (long)(qBase + row) * D + hcol + half * 32);
    uint4 x0 = p[0], x1 = p[1], x2 = p[2], x3 = p[3];
    uint4* q = (uint4*)(sQ + row * PQ + half * 32);
    q[0] = x0; q[1] = x1; q[2] = x2; q[3] = x3;
  }

  v8f oA[4];
#pragma unroll
  for (int j = 0; j < 4; ++j) oA[j] = vzero8();
  float rowM[8], rowL[8];
#pragma unroll
  for (int r = 0; r < 8; ++r) { rowM[r] = -INFINITY; rowL[r] = 0.f; }

  const int nkb = qBase / 64 + 2;  // causal: keys [0, qBase+128)
  for (int kb = 0; kb < nkb; ++kb) {
    const int k0 = kb * 64;
    __syncthreads();  // protect sK/sVt reuse
    {  // stage K block [64][64]
      const int key = tid >> 2, q4 = (tid & 3) * 16;
      const uint4* p = (const uint4*)(Kb + (long)(k0 + key) * D + hcol + q4);
      uint4 x0 = p[0], x1 = p[1];
      uint4* q = (uint4*)(sK + key * PQ + q4);
      q[0] = x0; q[1] = x1;
    }
    {  // stage V transposed: sVt[hd][key]
      const int key = tid & 63, hq = (tid >> 6) * 16;
      const unsigned short* p = Vb + (long)(k0 + key) * D + hcol + hq;
#pragma unroll
      for (int j = 0; j < 16; ++j) sVt[(hq + j) * PQ + key] = p[j];
    }
    __syncthreads();

    // scores = Q * K^T (contraction over hd=64)
    v8f sc[4];
#pragma unroll
    for (int j = 0; j < 4; ++j) sc[j] = vzero8();
#pragma unroll
    for (int ks = 0; ks < 64; ks += 32) {
      const unsigned short* pa = sQ + (mw + lr) * PQ + ks + hi * 8;
      v16bf af = frag_ld2(pa, pa + 16);
#pragma unroll
      for (int j = 0; j < 4; ++j) {
        const unsigned short* pb = sK + (j * 16 + lr) * PQ + ks + hi * 8;
        v16bf bfr = frag_ld2(pb, pb + 16);
        sc[j] = WMMA_BF16(af, bfr, sc[j]);
      }
    }

    // online softmax per row (row = mw + r + 8*hi)
#pragma unroll
    for (int r = 0; r < 8; ++r) {
      const int qIdx = qBase + mw + r + 8 * hi;
      float sv[4];
      float mx = -INFINITY;
#pragma unroll
      for (int j = 0; j < 4; ++j) {
        const int key = k0 + j * 16 + lr;
        float v = sc[j][r] * 0.125f;  // 1/sqrt(64)
        v = (key <= qIdx) ? v : -INFINITY;
        sv[j] = v;
        mx = fmaxf(mx, v);
      }
      mx = fmaxf(mx, __shfl_xor(mx, 1));
      mx = fmaxf(mx, __shfl_xor(mx, 2));
      mx = fmaxf(mx, __shfl_xor(mx, 4));
      mx = fmaxf(mx, __shfl_xor(mx, 8));
      const float mnew = fmaxf(rowM[r], mx);
      const float alpha = __expf(rowM[r] - mnew);
      float lsum = 0.f;
#pragma unroll
      for (int j = 0; j < 4; ++j) {
        float pv = __expf(sv[j] - mnew);
        lsum += pv;
        sP[(mw + r + 8 * hi) * PQ + j * 16 + lr] = f2bf(pv);
      }
      lsum += __shfl_xor(lsum, 1);
      lsum += __shfl_xor(lsum, 2);
      lsum += __shfl_xor(lsum, 4);
      lsum += __shfl_xor(lsum, 8);
      rowL[r] = rowL[r] * alpha + lsum;
      rowM[r] = mnew;
#pragma unroll
      for (int j = 0; j < 4; ++j) oA[j][r] *= alpha;
    }

    // O += P * V (contraction over 64 keys); per-wave sP region, LDS in-order
#pragma unroll
    for (int ks = 0; ks < 64; ks += 32) {
      const unsigned short* pa = sP + (mw + lr) * PQ + ks + hi * 8;
      v16bf af = frag_ld2(pa, pa + 16);
#pragma unroll
      for (int j = 0; j < 4; ++j) {
        const unsigned short* pb = sVt + (j * 16 + lr) * PQ + ks + hi * 8;
        v16bf bfr = frag_ld2(pb, pb + 16);
        oA[j] = WMMA_BF16(af, bfr, oA[j]);
      }
    }
  }

#pragma unroll
  for (int j = 0; j < 4; ++j)
#pragma unroll
    for (int r = 0; r < 8; ++r) {
      float v = oA[j][r] / rowL[r];
      Ob[(long)(qBase + mw + r + 8 * hi) * D + hcol + j * 16 + lr] = f2bf(v);
    }
}

// ---------------------------------------------------------------------------
// Router + routing bookkeeping
// ---------------------------------------------------------------------------
__global__ void router_kernel(const unsigned short* __restrict__ h2,
                              const float* __restrict__ Wr,
                              const float* __restrict__ br,
                              int* __restrict__ tokE, float* __restrict__ tokW,
                              int* __restrict__ counts, int S, int D) {
  const int t = blockIdx.x * blockDim.x + threadIdx.x;
  if (t >= S) return;
  float acc[8];
#pragma unroll
  for (int e = 0; e < 8; ++e) acc[e] = br[e];
  const unsigned short* hr = h2 + (long)t * D;
  for (int d = 0; d < D; ++d) {
    const float hv = bf2f(hr[d]);
    const float* wrow = Wr + d * 8;
#pragma unroll
    for (int e = 0; e < 8; ++e) acc[e] += hv * wrow[e];
  }
  int e0 = 0;
  float l0 = acc[0];
#pragma unroll
  for (int e = 1; e < 8; ++e)
    if (acc[e] > l0) { l0 = acc[e]; e0 = e; }
  int e1 = -1;
  float l1 = -INFINITY;
#pragma unroll
  for (int e = 0; e < 8; ++e)
    if (e != e0 && acc[e] > l1) { l1 = acc[e]; e1 = e; }
  const float p0 = 1.f / (1.f + __expf(l1 - l0));
  tokE[t * 2] = e0;
  tokE[t * 2 + 1] = e1;
  tokW[t * 2] = p0;
  tokW[t * 2 + 1] = 1.f - p0;
  atomicAdd(&counts[e0], 1);
  atomicAdd(&counts[e1], 1);
}

__global__ void init_kernel(int* counts) {
  if (threadIdx.x < 8) counts[threadIdx.x] = 0;
}

__global__ void scan_kernel(const int* __restrict__ counts, int* __restrict__ base,
                            int* __restrict__ cursor) {
  if (threadIdx.x == 0) {
    int b = 0;
    for (int e = 0; e < 8; ++e) {
      base[e] = b;
      cursor[e] = b;
      b += counts[e];
    }
    base[8] = b;
  }
}

__global__ void fill_kernel(const int* __restrict__ tokE, const float* __restrict__ tokW,
                            int* __restrict__ cursor, int* __restrict__ pairTok,
                            float* __restrict__ pairW, int* __restrict__ tokSlot, int S) {
  const int t = blockIdx.x * blockDim.x + threadIdx.x;
  if (t >= S) return;
  for (int j = 0; j < 2; ++j) {
    const int e = tokE[t * 2 + j];
    const int slot = atomicAdd(&cursor[e], 1);
    pairTok[slot] = t;
    pairW[slot] = tokW[t * 2 + j];
    tokSlot[t * 2 + j] = slot;
  }
}

// out = a + eo[slot0] + eo[slot1]   (routing weight already applied in GEMM2)
__global__ void reduce_kernel(const float* __restrict__ a, const float* __restrict__ eo,
                              const int* __restrict__ tokSlot, float* __restrict__ out,
                              int D, long total) {
  const long i = (long)blockIdx.x * 256 + threadIdx.x;
  if (i >= total) return;
  const int t = (int)(i / D);
  const int d = (int)(i % D);
  const long s0 = tokSlot[t * 2], s1 = tokSlot[t * 2 + 1];
  out[i] = a[i] + eo[s0 * D + d] + eo[s1 * D + d];
}

// ---------------------------------------------------------------------------
extern "C" void kernel_launch(void* const* d_in, const int* in_sizes, int n_in,
                              void* d_out, int out_size, void* d_ws, size_t ws_size,
                              hipStream_t stream) {
  (void)in_sizes; (void)n_in; (void)out_size; (void)ws_size;
  const int S = 2048, D = 1024, F = 4096, E = 8, H = 16, NP = 2 * S;

  const float* x    = (const float*)d_in[0];
  const float* ln1g = (const float*)d_in[1];
  const float* ln1b = (const float*)d_in[2];
  const float* ln2g = (const float*)d_in[3];
  const float* ln2b = (const float*)d_in[4];
  const float* Wq = (const float*)d_in[5];   const float* bq = (const float*)d_in[6];
  const float* Wk = (const float*)d_in[7];   const float* bk = (const float*)d_in[8];
  const float* Wv = (const float*)d_in[9];   const float* bv = (const float*)d_in[10];
  const float* Wo = (const float*)d_in[11];  const float* bo = (const float*)d_in[12];
  const float* Wr = (const float*)d_in[13];  const float* br = (const float*)d_in[14];
  const float* e_w1 = (const float*)d_in[15]; const float* e_b1 = (const float*)d_in[16];
  const float* e_w2 = (const float*)d_in[17]; const float* e_b2 = (const float*)d_in[18];
  float* outp = (float*)d_out;

  size_t off = 0;
  char* base0 = (char*)d_ws;
  auto carve = [&](size_t bytes) -> void* {
    void* p = base0 + off;
    off += (bytes + 255) & ~(size_t)255;
    return p;
  };
  unsigned short* h_bf  = (unsigned short*)carve((size_t)S * D * 2);
  unsigned short* q_bf  = (unsigned short*)carve((size_t)S * D * 2);
  unsigned short* k_bf  = (unsigned short*)carve((size_t)S * D * 2);
  unsigned short* v_bf  = (unsigned short*)carve((size_t)S * D * 2);
  unsigned short* o_bf  = (unsigned short*)carve((size_t)S * D * 2);
  float*          a_f   = (float*)carve((size_t)S * D * 4);
  unsigned short* h2_bf = (unsigned short*)carve((size_t)S * D * 2);
  unsigned short* mid   = (unsigned short*)carve((size_t)NP * F * 2);
  float*          eo    = (float*)carve((size_t)NP * D * 4);
  unsigned short* Wqt   = (unsigned short*)carve((size_t)D * D * 2);
  unsigned short* Wkt   = (unsigned short*)carve((size_t)D * D * 2);
  unsigned short* Wvt   = (unsigned short*)carve((size_t)D * D * 2);
  unsigned short* Wot   = (unsigned short*)carve((size_t)D * D * 2);
  unsigned short* w1t   = (unsigned short*)carve((size_t)E * D * F * 2);  // [E][F][D]
  unsigned short* w2t   = (unsigned short*)carve((size_t)E * F * D * 2);  // [E][D][F]
  int*   tokE    = (int*)carve((size_t)S * 2 * 4);
  float* tokW    = (float*)carve((size_t)S * 2 * 4);
  int*   tokSlot = (int*)carve((size_t)S * 2 * 4);
  int*   pairTok = (int*)carve((size_t)NP * 4);
  float* pairW   = (float*)carve((size_t)NP * 4);
  int*   counts  = (int*)carve(16 * 4);
  int*   segBase = (int*)carve(16 * 4);
  int*   cursor  = (int*)carve(16 * 4);

  init_kernel<<<1, 32, 0, stream>>>(counts);

  // one-time weight transpose + bf16 conversion (after this, all GEMM operands
  // are bf16 and K-contiguous; ~142 MB, resident in the 192 MB L2)
  dim3 gT(D / 32, D / 32, 1);
  transpose_cvt_kernel<<<gT, 256, 0, stream>>>(Wq, Wqt, D, D);
  transpose_cvt_kernel<<<gT, 256, 0, stream>>>(Wk, Wkt, D, D);
  transpose_cvt_kernel<<<gT, 256, 0, stream>>>(Wv, Wvt, D, D);
  transpose_cvt_kernel<<<gT, 256, 0, stream>>>(Wo, Wot, D, D);
  transpose_cvt_kernel<<<dim3(F / 32, D / 32, E), 256, 0, stream>>>(e_w1, w1t, D, F);
  transpose_cvt_kernel<<<dim3(D / 32, F / 32, E), 256, 0, stream>>>(e_w2, w2t, F, D);

  // LN1 -> h (bf16)
  ln_kernel<<<S, 256, 0, stream>>>(x, ln1g, ln1b, h_bf, D);

  // QKV projections
  dim3 gQ(D / TILE, S / TILE, 1);
  gemm_bf16_kernel<<<gQ, 256, 0, stream>>>(h_bf, D, Wqt, D, 0, bq, 0, nullptr,
                                           nullptr, nullptr, nullptr, nullptr,
                                           nullptr, q_bf, D, S, D, D, 0);
  gemm_bf16_kernel<<<gQ, 256, 0, stream>>>(h_bf, D, Wkt, D, 0, bk, 0, nullptr,
                                           nullptr, nullptr, nullptr, nullptr,
                                           nullptr, k_bf, D, S, D, D, 0);
  gemm_bf16_kernel<<<gQ, 256, 0, stream>>>(h_bf, D, Wvt, D, 0, bv, 0, nullptr,
                                           nullptr, nullptr, nullptr, nullptr,
                                           nullptr, v_bf, D, S, D, D, 0);

  // causal flash attention
  dim3 gA(S / 128, H, 1);
  attn_kernel<<<gA, 256, 0, stream>>>(q_bf, k_bf, v_bf, o_bf, S, D);

  // output projection + residual -> a (f32)
  gemm_bf16_kernel<<<gQ, 256, 0, stream>>>(o_bf, D, Wot, D, 0, bo, 0, x,
                                           nullptr, nullptr, nullptr, nullptr,
                                           a_f, nullptr, D, S, D, D, 1);

  // LN2 -> h2 (bf16)
  ln_kernel<<<S, 256, 0, stream>>>(a_f, ln2g, ln2b, h2_bf, D);

  // routing
  router_kernel<<<S / 256, 256, 0, stream>>>(h2_bf, Wr, br, tokE, tokW, counts, S, D);
  scan_kernel<<<1, 1, 0, stream>>>(counts, segBase, cursor);
  fill_kernel<<<S / 256, 256, 0, stream>>>(tokE, tokW, cursor, pairTok, pairW, tokSlot, S);

  // expert GEMM1: mid = gelu(gather(h2) @ e_w1 + e_b1)   [bf16]
  dim3 g1(F / TILE, S / TILE, E);
  gemm_bf16_kernel<<<g1, 256, 0, stream>>>(h2_bf, D, w1t, D, (long)F * D,
                                           e_b1, F, nullptr, segBase, counts,
                                           pairTok, nullptr, nullptr, mid, F,
                                           0, F, D, 2);

  // expert GEMM2: eo = (mid @ e_w2 + e_b2) * routeWeight   [f32]
  dim3 g2(D / TILE, S / TILE, E);
  gemm_bf16_kernel<<<g2, 256, 0, stream>>>(mid, F, w2t, F, (long)D * F,
                                           e_b2, D, nullptr, segBase, counts,
                                           nullptr, pairW, eo, nullptr, D,
                                           0, D, F, 3);

  // final: out = a + eo[slot0] + eo[slot1]
  const long total = (long)S * D;
  reduce_kernel<<<(unsigned)((total + 255) / 256), 256, 0, stream>>>(
      a_f, eo, tokSlot, outp, D, total);
}
